// GeometricContextExtractor_82781199663405
// MI455X (gfx1250) — matrix operand order repaired and checked by reading, not compiled, not run on hardware
//
#include <hip/hip_runtime.h>

// ---------------------------------------------------------------------------
// GeometricContextExtractor for MI455X (gfx1250, wave32, WMMA)
// B=2, N=8192, K=16 neighbors, S=256 segments, D=768
// Weights are pre-transposed to f16 so both A and B WMMA fragments load as
// contiguous 16-byte chunks (global_load_b128), no divergent guards in the
// hot GEMMs.
// ---------------------------------------------------------------------------

typedef _Float16 h8   __attribute__((ext_vector_type(8)));
typedef _Float16 v16h __attribute__((ext_vector_type(16)));
typedef float    v8f  __attribute__((ext_vector_type(8)));

#define NPTS   8192
#define NBATCH 2
#define BN     16384      // NBATCH*NPTS
#define KNN    16
#define NKROWS 262144     // BN*KNN
#define SEG    256
#define DIM    768

// ------------------------- WMMA fragment loaders ---------------------------
// A-matrix 16x32 f16 layout (ISA 7.12.2):
//   lanes 0-15 : row M=l, halves = K{kk+0..7, kk+16..23}
//   lanes16-31 : row M=l, halves = K{kk+8..15, kk+24..31}
template <bool G>
__device__ __forceinline__ v16h load_a_frag(const _Float16* A, int lda, int row,
                                            int kk, int kreal) {
    int lane = threadIdx.x & 31;
    int l = lane & 15, hi = lane >> 4;
    const _Float16* rp = A + (size_t)(row + l) * lda;
    int k0 = kk + (hi ? 8 : 0);
    int k1 = kk + 16 + (hi ? 8 : 0);
    h8 z = {0, 0, 0, 0, 0, 0, 0, 0};
    h8 c0 = (!G || k0 < kreal) ? *(const h8*)(rp + k0) : z;
    h8 c1 = (!G || k1 < kreal) ? *(const h8*)(rp + k1) : z;
    v16h a;
#pragma unroll
    for (int i = 0; i < 8; ++i) { a[i] = c0[i]; a[8 + i] = c1[i]; }
    return a;
}

// B-matrix 32x16 f16 fragment from TRANSPOSED weights Bt[N,K] (row-major):
//   lanes 0-15 : col N=l, halves = K{kk+0..7, kk+16..23}   -> Bt row (col+l)
//   lanes16-31 : col N=l, halves = K{kk+8..15, kk+24..31}
template <bool G>
__device__ __forceinline__ v16h load_bt_frag(const _Float16* Bt, int ldbt,
                                             int kk, int col, int kreal) {
    int lane = threadIdx.x & 31;
    int l = lane & 15, hi = lane >> 4;
    int kb = kk + hi * 8;
    const _Float16* cp = Bt + (size_t)(col + l) * ldbt + kb;
    h8 z = {0, 0, 0, 0, 0, 0, 0, 0};
    h8 b0 = (!G || kb < kreal)      ? *(const h8*)(cp)      : z;
    h8 b1 = (!G || kb + 16 < kreal) ? *(const h8*)(cp + 16) : z;
    v16h b;
#pragma unroll
    for (int i = 0; i < 8; ++i) { b[i] = b0[i]; b[8 + i] = b1[i]; }
    return b;
}

// --------------------------- generic WMMA GEMM -----------------------------
// C[M,Nc] = A[M,Kreal](f16) @ Bt[Nc,Kreal]^T(f16) + bias, optional ReLU.
// Each wave: one 16-row tile x NT adjacent 16-col tiles (A frag reused NT x).
template <bool G, int NT>
__global__ __launch_bounds__(256)
void wmma_gemm_kernel(const _Float16* __restrict__ A,
                      const _Float16* __restrict__ Bt,
                      const float* __restrict__ bias,
                      float* __restrict__ outF, _Float16* __restrict__ outH,
                      int M, int Nc, int Kreal,
                      int lda, int ldbt, int ldc, int relu) {
    int colGroups = Nc / (16 * NT);
    int groups = (M >> 4) * colGroups;
    int g = blockIdx.x * (blockDim.x >> 5) + (threadIdx.x >> 5);
    if (g >= groups) return;               // wave-uniform (EXEC all-1 for wmma)
    int rb  = (g / colGroups) << 4;
    int cb0 = (g % colGroups) * (16 * NT);

    v8f c[NT];
#pragma unroll
    for (int t = 0; t < NT; ++t) c[t] = (v8f){0.f, 0.f, 0.f, 0.f, 0.f, 0.f, 0.f, 0.f};

    int Kr = G ? ((Kreal + 31) & ~31) : Kreal;
    for (int kk = 0; kk < Kr; kk += 32) {
        v16h a = load_a_frag<G>(A, lda, rb, kk, Kreal);
#pragma unroll
        for (int t = 0; t < NT; ++t) {
            v16h b = load_bt_frag<G>(Bt, ldbt, kk, cb0 + 16 * t, Kreal);
            c[t] = __builtin_amdgcn_wmma_f32_16x16x32_f16(false, a, false, b,
                                                          (short)0, c[t], false, false);
        }
    }
    int lane = threadIdx.x & 31;
    int l = lane & 15, hi = lane >> 4;
#pragma unroll
    for (int t = 0; t < NT; ++t) {
        int col = cb0 + 16 * t + l;
        float bv = bias ? bias[col] : 0.f;
#pragma unroll
        for (int v = 0; v < 8; ++v) {      // C layout: VGPR v -> row hi*8+v, col l
            int row = rb + hi * 8 + v;
            float val = c[t][v] + bv;
            if (relu) val = fmaxf(val, 0.f);
            if (outF) outF[(size_t)row * ldc + col] = val;
            else      outH[(size_t)row * ldc + col] = (_Float16)val;
        }
    }
}

// ---------------- GEMM2 (128->256) + ReLU + mean over 16 neighbors ---------
// Row tile = the 16 neighbor rows of one point -> one h2bar row per tile.
__global__ __launch_bounds__(256)
void gemm2_mean_kernel(const _Float16* __restrict__ h1,
                       const _Float16* __restrict__ W2t,   // [256,128]
                       const float* __restrict__ b2,
                       _Float16* __restrict__ h2bar) {
    int tile = blockIdx.x * (blockDim.x >> 5) + (threadIdx.x >> 5);
    if (tile >= BN * 16) return;
    int p  = tile >> 4;            // point id 0..16383
    int cb = (tile & 15) << 4;     // 256 cols -> 16 col tiles
    int rb = p << 4;

    v8f c = {0.f, 0.f, 0.f, 0.f, 0.f, 0.f, 0.f, 0.f};
    for (int kk = 0; kk < 128; kk += 32) {
        v16h a = load_a_frag<false>(h1, 128, rb, kk, 128);
        v16h b = load_bt_frag<false>(W2t, 128, kk, cb, 128);
        c = __builtin_amdgcn_wmma_f32_16x16x32_f16(false, a, false, b,
                                                   (short)0, c, false, false);
    }
    int lane = threadIdx.x & 31;
    int l = lane & 15;
    float bv = b2[cb + l];
    float s = 0.f;
#pragma unroll
    for (int v = 0; v < 8; ++v) {          // relu THEN mean (matches reference)
        float x = c[v] + bv;
        s += fmaxf(x, 0.f);
    }
    s += __shfl_xor(s, 16);                // combine rows 0-7 with 8-15
    if (lane < 16)
        h2bar[(size_t)p * 256 + cb + l] = (_Float16)(s * (1.f / 16.f));
}

// ----------------------------- KNN + features ------------------------------
// One wave per query point. Candidates streamed through LDS tiles of 1024.
// Per-lane running top-16 (squared distances), butterfly-shuffle merge.
__global__ __launch_bounds__(256)
void knn_feat_kernel(const float* __restrict__ coords,
                     _Float16* __restrict__ feat) {
    __shared__ float sx[1024], sy[1024], sz[1024];
    int w = threadIdx.x >> 5;
    int lane = threadIdx.x & 31;
    int p = blockIdx.x * 8 + w;            // 0..16383, block never straddles batches
    int b = p >> 13;
    const float* cb0 = coords + (size_t)b * NPTS * 3;
    const float* cp  = coords + (size_t)p * 3;
    float px = cp[0], py = cp[1], pz = cp[2];

    float bd[16]; int bi[16];
#pragma unroll
    for (int t = 0; t < 16; ++t) { bd[t] = 3.0e38f; bi[t] = 0; }
    float cmax = 3.0e38f;

    for (int tile = 0; tile < NPTS / 1024; ++tile) {
        __syncthreads();
        for (int j = threadIdx.x; j < 1024; j += 256) {
            const float* q = cb0 + (size_t)(tile * 1024 + j) * 3;
            sx[j] = q[0]; sy[j] = q[1]; sz[j] = q[2];
        }
        __syncthreads();
        for (int ii = 0; ii < 32; ++ii) {
            int j = lane + (ii << 5);
            float dx = sx[j] - px, dy = sy[j] - py, dz = sz[j] - pz;
            float d2 = dx * dx + dy * dy + dz * dz;
            if (d2 < cmax) {               // rare after warm-up
                int slot = 0; float m = bd[0];
#pragma unroll
                for (int t = 1; t < 16; ++t) if (bd[t] > m) { m = bd[t]; slot = t; }
                bd[slot] = d2; bi[slot] = tile * 1024 + j;
                cmax = bd[0];
#pragma unroll
                for (int t = 1; t < 16; ++t) cmax = fmaxf(cmax, bd[t]);
            }
        }
    }

    // sort each lane's 16 ascending
#pragma unroll
    for (int i = 1; i < 16; ++i) {
        float v = bd[i]; int ix = bi[i]; int t = i - 1;
        while (t >= 0 && bd[t] > v) { bd[t + 1] = bd[t]; bi[t + 1] = bi[t]; --t; }
        bd[t + 1] = v; bi[t + 1] = ix;
    }

    // 16-round merge of 32 sorted lists: global min via wave32 butterfly
    int head = 0, nb = 0;
    for (int r = 0; r < 16; ++r) {
        float v = (head < 16) ? bd[head] : 3.0e38f;
        float bv = v; int bl = lane;
#pragma unroll
        for (int off = 16; off; off >>= 1) {
            float ov = __shfl_xor(bv, off);
            int   ol = __shfl_xor(bl, off);
            if (ov < bv || (ov == bv && ol < bl)) { bv = ov; bl = ol; }
        }
        int hc = (head < 16) ? head : 15;
        int widx = __shfl(bi[hc], bl);
        if (lane == r)  nb = widx;         // neighbor r owned by lane r
        if (lane == bl) head++;
    }

    if (lane < KNN) {
        const float* q = cb0 + (size_t)nb * 3;
        float rx = q[0] - px, ry = q[1] - py, rz = q[2] - pz;
        float rd = sqrtf(rx * rx + ry * ry + rz * rz);
        float inv = 1.0f / (rd + 1e-6f);
        float nx = rx * inv, ny = ry * inv, nz = rz * inv;
        float axy = atan2f(ny, nx);
        float axz = atan2f(nz, nx);
        float ayz = atan2f(nz, ny);
        _Float16* fr = feat + ((size_t)p * KNN + lane) * 8;
        fr[0] = (_Float16)rd;  fr[1] = (_Float16)rx;  fr[2] = (_Float16)ry;
        fr[3] = (_Float16)rz;  fr[4] = (_Float16)axy; fr[5] = (_Float16)axz;
        fr[6] = (_Float16)ayz; fr[7] = (_Float16)0.f;
    }
}

// ----------------------------- small helpers -------------------------------
// Transpose-convert: src[rows,cols] f32 -> dst[cols,ldd] f16 (dst[c][r]=src[r][c],
// zero-padded for r >= rows). ldd multiple of 8 keeps fragments 16B-aligned.
__global__ void transpose_to_f16_kernel(const float* __restrict__ src,
                                        _Float16* __restrict__ dst,
                                        int rows, int cols, int ldd) {
    int i = blockIdx.x * blockDim.x + threadIdx.x;
    if (i >= cols * ldd) return;
    int c = i / ldd, r = i % ldd;
    dst[i] = (r < rows) ? (_Float16)src[(size_t)r * cols + c] : (_Float16)0.f;
}
__global__ void zero_f32_kernel(float* p, int n) {
    int i = blockIdx.x * blockDim.x + threadIdx.x;
    if (i < n) p[i] = 0.f;
}

__global__ __launch_bounds__(256)
void segsum_kernel(const float* __restrict__ pc, const int* __restrict__ labels,
                   float* __restrict__ sums, float* __restrict__ counts) {
    int p = blockIdx.x;                    // one block per point
    int lab = labels[p];
    int b = p >> 13;
    float* srow = sums + ((size_t)(b * SEG + lab)) * DIM;
    const float* prow = pc + (size_t)p * DIM;
    for (int d = threadIdx.x; d < DIM; d += blockDim.x)
        atomicAdd(&srow[d], prow[d]);
    if (threadIdx.x == 0) atomicAdd(&counts[b * SEG + lab], 1.0f);
}

__global__ void means_kernel(const float* __restrict__ sums,
                             const float* __restrict__ counts,
                             _Float16* __restrict__ mh) {
    int i = blockIdx.x * blockDim.x + threadIdx.x;
    if (i >= NBATCH * SEG * DIM) return;
    int s = i / DIM;
    float c = fmaxf(counts[s], 1.0f);
    mh[i] = (_Float16)(sums[i] / c);
}

__global__ __launch_bounds__(256)
void ln_relu_kernel(const float* __restrict__ a, const float* __restrict__ g,
                    const float* __restrict__ bb, _Float16* __restrict__ outh) {
    __shared__ float red[256];
    int row = blockIdx.x;
    const float* ar = a + (size_t)row * DIM;
    float s = 0.f, s2 = 0.f;
    for (int d = threadIdx.x; d < DIM; d += 256) {
        float x = ar[d]; s += x; s2 += x * x;
    }
    red[threadIdx.x] = s; __syncthreads();
    for (int o = 128; o; o >>= 1) {
        if (threadIdx.x < o) red[threadIdx.x] += red[threadIdx.x + o];
        __syncthreads();
    }
    float S = red[0]; __syncthreads();
    red[threadIdx.x] = s2; __syncthreads();
    for (int o = 128; o; o >>= 1) {
        if (threadIdx.x < o) red[threadIdx.x] += red[threadIdx.x + o];
        __syncthreads();
    }
    float S2 = red[0];
    float mu = S / (float)DIM;
    float var = S2 / (float)DIM - mu * mu;  // ddof=0 like jnp.var
    float inv = rsqrtf(var + 1e-5f);
    for (int d = threadIdx.x; d < DIM; d += 256) {
        float y = (ar[d] - mu) * inv * g[d] + bb[d];
        outh[(size_t)row * DIM + d] = (_Float16)fmaxf(y, 0.f);
    }
}

__global__ void blend_kernel(const float* __restrict__ pc,
                             const float* __restrict__ agg,
                             const int* __restrict__ labels,
                             const float* __restrict__ counts,
                             float* __restrict__ out) {
    size_t i = (size_t)blockIdx.x * blockDim.x + threadIdx.x;
    if (i >= (size_t)BN * DIM) return;
    int p = (int)(i / DIM);
    int d = (int)(i % DIM);
    int b = p >> 13;
    int lab = labels[p];
    float pcv = pc[i];
    float o = pcv;
    if (counts[b * SEG + lab] >= 2.0f)
        o = 0.8f * pcv + 0.2f * agg[((size_t)(b * SEG + lab)) * DIM + d];
    out[i] = o;
}

// ------------------------------- launcher ----------------------------------
extern "C" void kernel_launch(void* const* d_in, const int* in_sizes, int n_in,
                              void* d_out, int out_size, void* d_ws, size_t ws_size,
                              hipStream_t stream) {
    (void)in_sizes; (void)n_in; (void)out_size; (void)ws_size;
    const float* coords = (const float*)d_in[0];
    const int*   labels = (const int*)d_in[1];
    const float* W1 = (const float*)d_in[2];  const float* b1 = (const float*)d_in[3];
    const float* W2 = (const float*)d_in[4];  const float* b2 = (const float*)d_in[5];
    const float* W3 = (const float*)d_in[6];  const float* b3 = (const float*)d_in[7];
    const float* Wa = (const float*)d_in[8];  const float* ba = (const float*)d_in[9];
    const float* lg = (const float*)d_in[10]; const float* lb = (const float*)d_in[11];
    const float* Wb = (const float*)d_in[12]; const float* bb = (const float*)d_in[13];
    float* out = (float*)d_out;

    uint8_t* base = (uint8_t*)d_ws;
    size_t off = 0;
    auto take = [&](size_t bytes) -> void* {
        void* r = base + off;
        off += (bytes + 255) & ~(size_t)255;
        return r;
    };
    _Float16* feat = (_Float16*)take((size_t)NKROWS * 8 * 2);
    _Float16* W1t  = (_Float16*)take(128 * 8 * 2);               // [128,8]
    _Float16* W2t  = (_Float16*)take(256 * 128 * 2);             // [256,128]
    _Float16* W3t  = (_Float16*)take((size_t)DIM * 256 * 2);     // [768,256]
    _Float16* Wat  = (_Float16*)take((size_t)DIM * DIM * 2);     // [768,768]
    _Float16* Wbt  = (_Float16*)take((size_t)DIM * DIM * 2);     // [768,768]
    _Float16* h1   = (_Float16*)take((size_t)NKROWS * 128 * 2);
    _Float16* h2b  = (_Float16*)take((size_t)BN * 256 * 2);
    float*    pc   = (float*)take((size_t)BN * DIM * 4);
    float*    sums = (float*)take((size_t)NBATCH * SEG * DIM * 4);
    float*    cnts = (float*)take((size_t)NBATCH * SEG * 4);
    _Float16* mh   = (_Float16*)take((size_t)NBATCH * SEG * DIM * 2);
    float*    abuf = (float*)take((size_t)NBATCH * SEG * DIM * 4);
    _Float16* ah   = (_Float16*)take((size_t)NBATCH * SEG * DIM * 2);
    float*    agg  = (float*)take((size_t)NBATCH * SEG * DIM * 4);

    // weights -> transposed f16 (zero-padded K)
    transpose_to_f16_kernel<<<(128 * 8 + 255) / 256, 256, 0, stream>>>(W1, W1t, 7, 128, 8);
    transpose_to_f16_kernel<<<(256 * 128 + 255) / 256, 256, 0, stream>>>(W2, W2t, 128, 256, 128);
    transpose_to_f16_kernel<<<(DIM * 256 + 255) / 256, 256, 0, stream>>>(W3, W3t, 256, DIM, 256);
    transpose_to_f16_kernel<<<(DIM * DIM + 255) / 256, 256, 0, stream>>>(Wa, Wat, DIM, DIM, DIM);
    transpose_to_f16_kernel<<<(DIM * DIM + 255) / 256, 256, 0, stream>>>(Wb, Wbt, DIM, DIM, DIM);
    zero_f32_kernel<<<(NBATCH * SEG * DIM + 255) / 256, 256, 0, stream>>>(sums, NBATCH * SEG * DIM);
    zero_f32_kernel<<<(NBATCH * SEG + 255) / 256, 256, 0, stream>>>(cnts, NBATCH * SEG);

    // KNN + 7-dim geometric features
    knn_feat_kernel<<<BN / 8, 256, 0, stream>>>(coords, feat);

    // GEMM1 (guarded, K=8): [NK,8] @ [8,128]^T + b1, ReLU -> h1 (f16)
    //   groups = 16384 rowTiles * (128/64) = 32768 -> 4096 blocks
    wmma_gemm_kernel<true, 4><<<4096, 256, 0, stream>>>(feat, W1t, b1, nullptr, h1,
                                                        NKROWS, 128, 8, 8, 8, 128, 1);
    // GEMM2 + ReLU + mean over 16 neighbors -> h2bar [BN,256] (262144 tiles)
    gemm2_mean_kernel<<<32768, 256, 0, stream>>>(h1, W2t, b2, h2b);
    // GEMM3: [BN,256] @ [256,768]^T + b3 -> pc (f32)
    //   groups = 1024 * (768/64) = 12288 -> 1536 blocks
    wmma_gemm_kernel<false, 4><<<1536, 256, 0, stream>>>(h2b, W3t, b3, pc, nullptr,
                                                         BN, DIM, 256, 256, 256, DIM, 0);
    // superpoint segment sums + counts
    segsum_kernel<<<BN, 256, 0, stream>>>(pc, labels, sums, cnts);
    means_kernel<<<(NBATCH * SEG * DIM + 255) / 256, 256, 0, stream>>>(sums, cnts, mh);
    // aggregator: Linear -> LN -> ReLU -> Linear  (512x768 rows)
    //   groups = 32 * 12 = 384 -> 48 blocks
    wmma_gemm_kernel<false, 4><<<48, 256, 0, stream>>>(mh, Wat, ba, abuf, nullptr,
                                                       NBATCH * SEG, DIM, DIM, DIM, DIM, DIM, 0);
    ln_relu_kernel<<<NBATCH * SEG, 256, 0, stream>>>(abuf, lg, lb, ah);
    wmma_gemm_kernel<false, 4><<<48, 256, 0, stream>>>(ah, Wbt, bb, agg, nullptr,
                                                       NBATCH * SEG, DIM, DIM, DIM, DIM, DIM, 0);
    // final consistency blend
    blend_kernel<<<(BN * DIM + 255) / 256, 256, 0, stream>>>(pc, agg, labels, cnts, out);
}
